// LSTMTagCRF_10488310136835
// MI455X (gfx1250) — compile-verified, matrix operand order checked by reading, not run on hardware
//
#include <hip/hip_runtime.h>
#include <hip/hip_bf16.h>

// ---------------------------------------------------------------------------
// Problem constants
// ---------------------------------------------------------------------------
#define BB   64
#define SS   512
#define DD   256
#define HH   512
#define TT   32
#define MM   (BB * SS)          // 32768 rows
#define G4   (4 * HH)           // 2048 gate cols per direction
#define N1   (2 * G4)           // 4096 (both directions concatenated)

// LDS staging of Whh: 4 waves/block x 64 rows x (512 + 8 pad) bf16
#define LROWS 64
#define LSTR  (HH + 8)          // 520 elems = 1040 B; 1040 % 256 == 16 -> bank spread
#define LDS_BYTES (4 * LROWS * LSTR * 2)   // 266240 B <= 320 KB WGP LDS

#if defined(__AMDGCN__) && __has_builtin(__builtin_amdgcn_global_load_async_to_lds_b128)
#define USE_ASYNC_STAGE 1
#endif

typedef __attribute__((ext_vector_type(16))) __bf16 bf16x16;
typedef __attribute__((ext_vector_type(8)))  float  f32x8;
typedef int nv_i4 __attribute__((vector_size(16)));   // native vector for builtins

// ---------------------------------------------------------------------------
// WMMA operand tile load.
// A (16x32 bf16, row-major [16,K-stride]) and B given as B^T ([N,K] row-major)
// share the same per-lane layout:
//   lane L: row/col = L&15, k-group = L>>4
//   elems 0..7  <- K = k0 + 8*kg .. +7        (16 bytes)
//   elems 8..15 <- K = k0 + 16 + 8*kg .. +7   (16 bytes)
// ---------------------------------------------------------------------------
union TileU { bf16x16 v; nv_i4 q[2]; };

__device__ __forceinline__ bf16x16 load_tile(const __bf16* __restrict__ p) {
  TileU t;
  t.q[0] = *reinterpret_cast<const nv_i4*>(p);
  t.q[1] = *reinterpret_cast<const nv_i4*>(p + 16);
  return t.v;
}

__device__ __forceinline__ f32x8 vzero8() {
  f32x8 v;
#pragma unroll
  for (int i = 0; i < 8; ++i) v[i] = 0.0f;
  return v;
}

__device__ __forceinline__ float sigmoidf_(float x) {
  return 1.0f / (1.0f + expf(-x));
}

// ---------------------------------------------------------------------------
// 0) init: zero output scalar, grid-barrier counter, h-state double buffers
// ---------------------------------------------------------------------------
__global__ void init_kernel(float* __restrict__ out, unsigned int* __restrict__ ctr,
                            __bf16* __restrict__ hstate, int nh) {
  int i = blockIdx.x * blockDim.x + threadIdx.x;
  if (i == 0) { out[0] = 0.0f; ctr[0] = 0u; }
  if (i < nh) hstate[i] = (__bf16)0.0f;
}

// ---------------------------------------------------------------------------
// 1) f32 -> bf16 convert (weights)
// ---------------------------------------------------------------------------
__global__ void cvt_kernel(const float* __restrict__ src, __bf16* __restrict__ dst, int n) {
  int i = blockIdx.x * blockDim.x + threadIdx.x;
  if (i < n) dst[i] = (__bf16)src[i];
}

// ---------------------------------------------------------------------------
// 2) embedding gather + bf16 convert: x[b,s,:] = emb[sentence[b,s],:]
// ---------------------------------------------------------------------------
__global__ void embed_kernel(const int* __restrict__ sent, const float* __restrict__ emb,
                             __bf16* __restrict__ x) {
  int i = blockIdx.x * blockDim.x + threadIdx.x;   // over MM * (DD/4)
  if (i >= MM * (DD / 4)) return;
  int m  = i >> 6;            // DD/4 == 64
  int d4 = (i & 63) << 2;
  int tok = sent[m];
  float4 v = *reinterpret_cast<const float4*>(emb + (size_t)tok * DD + d4);
  __bf16* dst = x + (size_t)m * DD + d4;
  dst[0] = (__bf16)v.x; dst[1] = (__bf16)v.y;
  dst[2] = (__bf16)v.z; dst[3] = (__bf16)v.w;
}

// ---------------------------------------------------------------------------
// 3) Input projection GEMM:  xw[M,4096] = x[M,256] * Wih^T  (+ bias)
//    Wih bf16 is [4096,256] = rows of w_ih_f then w_ih_b (already [N,K]).
//    One wave computes one 16-row M tile x four 16-col N tiles.
// ---------------------------------------------------------------------------
__global__ void gemm_xw_kernel(const __bf16* __restrict__ X, const __bf16* __restrict__ Wih,
                               const float* __restrict__ bf, const float* __restrict__ bb,
                               float* __restrict__ xw) {
  const int tid  = threadIdx.x;
  const int lane = tid & 31;
  const int gw   = blockIdx.x * (blockDim.x >> 5) + (tid >> 5);
  const int mt   = gw >> 6;      // 0..2047
  const int ng   = gw & 63;      // 4 N-tiles each
  const int nlo  = lane & 15;
  const int khi  = lane >> 4;
  if (mt >= MM / 16) return;

  const __bf16* arow = X + (size_t)(mt * 16 + nlo) * DD + 8 * khi;
  f32x8 acc[4];
#pragma unroll
  for (int j = 0; j < 4; ++j) acc[j] = vzero8();

#pragma unroll
  for (int k0 = 0; k0 < DD; k0 += 32) {
    __builtin_prefetch(arow + k0 + 64, 0, 1);
    bf16x16 a = load_tile(arow + k0);
#pragma unroll
    for (int j = 0; j < 4; ++j) {
      const int brow = (ng * 4 + j) * 16 + nlo;
      bf16x16 b = load_tile(Wih + (size_t)brow * DD + k0 + 8 * khi);
      acc[j] = __builtin_amdgcn_wmma_f32_16x16x32_bf16(false, a, false, b,
                                                       (short)0, acc[j], false, false);
    }
  }

#pragma unroll
  for (int j = 0; j < 4; ++j) {
    const int n = (ng * 4 + j) * 16 + nlo;
    const float bias = (n < G4) ? bf[n] : bb[n - G4];
#pragma unroll
    for (int r = 0; r < 8; ++r) {
      const int m = mt * 16 + r + 8 * khi;
      xw[(size_t)m * N1 + n] = acc[j][r] + bias;
    }
  }
}

// ---------------------------------------------------------------------------
// grid-wide barrier (all 64 blocks resident; monotonic counter)
// ---------------------------------------------------------------------------
__device__ __forceinline__ void grid_barrier(unsigned int* ctr, unsigned int target) {
  __threadfence();
  __syncthreads();
  if (threadIdx.x == 0) {
    atomicAdd(ctr, 1u);
    while (atomicAdd(ctr, 0u) < target) { __builtin_amdgcn_s_sleep(1); }
  }
  __syncthreads();
  __threadfence();
}

// ---------------------------------------------------------------------------
// 4) Persistent bidirectional LSTM recurrence.
//    256 waves: dir(2) x row-tile(4, 16 batch rows) x h-tile(32, 16 h units).
//    A block's 4 waves share (dir, row-tile); each wave owns 16 h-units.
//    The block's Whh slice (256 rows x 512 K = 256 KB) is staged ONCE into
//    CDNA5's 320 KB WGP LDS via async global->LDS copies (row-padded against
//    bank conflicts); the 512 sequential steps then read B via ds_load_b128.
//    Cell state c lives in registers; bf16 h double-buffered in global for
//    cross-block visibility; per-row time reversal handles the backward dir.
// ---------------------------------------------------------------------------
__global__ void lstm_kernel(const float* __restrict__ xw, const __bf16* __restrict__ Whh,
                            const int* __restrict__ lengths,
                            __bf16* __restrict__ hstate,   // [2 dir][2 phase][64][512]
                            __bf16* __restrict__ h_all,    // [64][512][1024]
                            unsigned int* __restrict__ ctr) {
  extern __shared__ __bf16 lds_whh[];                      // [4][LROWS][LSTR]

  const int tid  = threadIdx.x;
  const int lane = tid & 31;
  const int widx = tid >> 5;                               // wave in block
  const int wave = blockIdx.x * (blockDim.x >> 5) + widx;  // 0..255
  const int dir  = wave >> 7;
  const int rt   = (wave >> 5) & 3;
  const int ct   = wave & 31;
  const int nlo  = lane & 15;
  const int khi  = lane >> 4;
  const int hcol = ct * 16 + nlo;

  const __bf16* WhhD = Whh + (size_t)dir * G4 * HH;
  __bf16* lB = lds_whh + (size_t)widx * LROWS * LSTR;

  // ---- stage this wave's 64 Whh rows (4 gates x 16 cols x K=512) into LDS ----
  for (int row = 0; row < LROWS; ++row) {
    const int g   = row >> 4;
    const int col = row & 15;
    const __bf16* src = WhhD + (size_t)(g * HH + ct * 16 + col) * HH;
    __bf16*       dst = lB + (size_t)row * LSTR;
#pragma unroll
    for (int kc = lane; kc < HH / 8; kc += 32) {           // 16-byte chunks
#ifdef USE_ASYNC_STAGE
      __builtin_amdgcn_global_load_async_to_lds_b128(
          (nv_i4*)(const_cast<__bf16*>(src) + kc * 8),
          (nv_i4*)(dst + kc * 8), 0, 0);
#else
      *reinterpret_cast<nv_i4*>(dst + kc * 8) =
          *reinterpret_cast<const nv_i4*>(src + kc * 8);
#endif
    }
  }
#ifdef USE_ASYNC_STAGE
#if __has_builtin(__builtin_amdgcn_s_wait_asynccnt)
  __builtin_amdgcn_s_wait_asynccnt(0);
#else
  asm volatile("s_wait_asynccnt 0x0" ::: "memory");
#endif
#endif
  __syncthreads();

  int b_r[8], len_r[8];
#pragma unroll
  for (int r = 0; r < 8; ++r) {
    b_r[r]   = rt * 16 + r + 8 * khi;
    len_r[r] = lengths[b_r[r]];
  }
  float c_r[8];
#pragma unroll
  for (int r = 0; r < 8; ++r) c_r[r] = 0.0f;

  __bf16* hs_base     = hstate + (size_t)dir * 2 * BB * HH;
  const unsigned nblk = gridDim.x;

  for (int t = 0; t < SS; ++t) {
    const __bf16* hs_rd = hs_base + (size_t)(t & 1) * BB * HH;
    __bf16*       hs_wr = hs_base + (size_t)((t + 1) & 1) * BB * HH;

    int tt_r[8];
    size_t xoff_r[8];
#pragma unroll
    for (int r = 0; r < 8; ++r) {
      const int L  = len_r[r];
      const int tt = (dir == 0) ? t : ((t < L) ? (L - 1 - t) : t);
      tt_r[r]   = tt;
      xoff_r[r] = ((size_t)b_r[r] * SS + tt) * N1 + (size_t)dir * G4;
    }

    // C init from precomputed input projection
    f32x8 acc[4];
#pragma unroll
    for (int g = 0; g < 4; ++g) {
#pragma unroll
      for (int r = 0; r < 8; ++r)
        acc[g][r] = xw[xoff_r[r] + g * HH + hcol];
    }

    // acc += h_prev * Whh^T   (A from global h-state, B from LDS)
    const __bf16* arow = hs_rd + (size_t)(rt * 16 + nlo) * HH + 8 * khi;
#pragma unroll 4
    for (int k0 = 0; k0 < HH; k0 += 32) {
      bf16x16 a = load_tile(arow + k0);
#pragma unroll
      for (int g = 0; g < 4; ++g) {
        bf16x16 b = load_tile(lB + (size_t)(g * 16 + nlo) * LSTR + k0 + 8 * khi);
        acc[g] = __builtin_amdgcn_wmma_f32_16x16x32_bf16(false, a, false, b,
                                                         (short)0, acc[g], false, false);
      }
    }

    // LSTM cell update (i,f,g,o gate order)
#pragma unroll
    for (int r = 0; r < 8; ++r) {
      const float ig = sigmoidf_(acc[0][r]);
      const float fg = sigmoidf_(acc[1][r]);
      const float gg = tanhf(acc[2][r]);
      const float og = sigmoidf_(acc[3][r]);
      const float c  = fg * c_r[r] + ig * gg;
      c_r[r] = c;
      const float h = og * tanhf(c);
      const __bf16 hb = (__bf16)h;
      hs_wr[(size_t)b_r[r] * HH + hcol] = hb;
      h_all[((size_t)b_r[r] * SS + tt_r[r]) * (2 * HH) + (size_t)dir * HH + hcol] = hb;
    }

    grid_barrier(ctr, (unsigned)(t + 1) * nblk);
  }
}

// ---------------------------------------------------------------------------
// 5) Emissions GEMM: em[M,32] = h_all[M,1024] * Wout^T + b_out
//    One wave per 16-row M tile; T=32 -> two 16-col N tiles.
// ---------------------------------------------------------------------------
__global__ void emis_kernel(const __bf16* __restrict__ Hall, const __bf16* __restrict__ Wout,
                            const float* __restrict__ bout, float* __restrict__ em) {
  const int tid  = threadIdx.x;
  const int lane = tid & 31;
  const int mt   = blockIdx.x * (blockDim.x >> 5) + (tid >> 5);
  if (mt >= MM / 16) return;
  const int nlo = lane & 15;
  const int khi = lane >> 4;

  f32x8 acc[2];
  acc[0] = vzero8(); acc[1] = vzero8();
  const __bf16* arow = Hall + (size_t)(mt * 16 + nlo) * (2 * HH) + 8 * khi;

#pragma unroll 4
  for (int k0 = 0; k0 < 2 * HH; k0 += 32) {
    __builtin_prefetch(arow + k0 + 128, 0, 1);
    bf16x16 a = load_tile(arow + k0);
#pragma unroll
    for (int j = 0; j < 2; ++j) {
      bf16x16 b = load_tile(Wout + (size_t)(j * 16 + nlo) * (2 * HH) + k0 + 8 * khi);
      acc[j] = __builtin_amdgcn_wmma_f32_16x16x32_bf16(false, a, false, b,
                                                       (short)0, acc[j], false, false);
    }
  }

#pragma unroll
  for (int j = 0; j < 2; ++j) {
    const int n = j * 16 + nlo;
    const float bias = bout[n];
#pragma unroll
    for (int r = 0; r < 8; ++r)
      em[(size_t)(mt * 16 + r + 8 * khi) * TT + n] = acc[j][r] + bias;
  }
}

// ---------------------------------------------------------------------------
// 6) CRF negative log-likelihood. One wave32 per batch row; lane j = tag j.
// ---------------------------------------------------------------------------
__global__ void crf_kernel(const float* __restrict__ em, const int* __restrict__ labels,
                           const int* __restrict__ lengths,
                           const float* __restrict__ start_t, const float* __restrict__ end_t,
                           const float* __restrict__ trans, float* __restrict__ out) {
  const int b = blockIdx.x;
  const int j = threadIdx.x;                 // 0..31
  const float* E    = em + (size_t)b * SS * TT;
  const int*   tags = labels + b * SS;
  const int    len  = lengths[b];

  // transition column j, kept in registers (fully unrolled accesses)
  float trc[TT];
#pragma unroll
  for (int i = 0; i < TT; ++i) trc[i] = trans[i * TT + j];

  // ---- numerator (gold path score), lane-parallel over t ----
  float sc = 0.0f;
  for (int t = 1 + j; t < SS; t += 32) {
    if (t < len)
      sc += trans[tags[t - 1] * TT + tags[t]] + E[t * TT + tags[t]];
  }
#pragma unroll
  for (int off = 16; off > 0; off >>= 1) sc += __shfl_xor(sc, off);

  // ---- denominator: forward algorithm, alpha[j] in lane j ----
  float alpha = start_t[j] + E[j];
  for (int t = 1; t < SS; ++t) {
    if (t >= len) break;                     // alpha frozen past length
    const float emj = E[t * TT + j];
    float m = -3.4e38f;
#pragma unroll
    for (int i = 0; i < TT; ++i) {
      const float v = __shfl(alpha, i) + trc[i];
      m = fmaxf(m, v);
    }
    float s = 0.0f;
#pragma unroll
    for (int i = 0; i < TT; ++i) {
      const float v = __shfl(alpha, i) + trc[i];
      s += expf(v - m);
    }
    alpha = m + logf(s) + emj;
  }

  float v = alpha + end_t[j];
  float m = v;
#pragma unroll
  for (int off = 16; off > 0; off >>= 1) m = fmaxf(m, __shfl_xor(m, off));
  float e = expf(v - m);
#pragma unroll
  for (int off = 16; off > 0; off >>= 1) e += __shfl_xor(e, off);
  const float den = m + logf(e);

  if (j == 0) {
    const float score = sc + start_t[tags[0]] + E[tags[0]] + end_t[tags[len - 1]];
    atomicAdd(out, den - score);
  }
}

// ---------------------------------------------------------------------------
// Workspace layout (bytes)
// ---------------------------------------------------------------------------
static constexpr size_t OFF_XW   = 0;                                      // f32 [M,4096]
static constexpr size_t OFF_X    = OFF_XW   + (size_t)MM * N1 * 4;         // bf16 [M,256]
static constexpr size_t OFF_HA   = OFF_X    + (size_t)MM * DD * 2;         // bf16 [M,1024]
static constexpr size_t OFF_WIH  = OFF_HA   + (size_t)MM * 2 * HH * 2;     // bf16 [4096,256]
static constexpr size_t OFF_WHH  = OFF_WIH  + (size_t)N1 * DD * 2;         // bf16 [2][2048,512]
static constexpr size_t OFF_WOUT = OFF_WHH  + (size_t)2 * G4 * HH * 2;     // bf16 [32,1024]
static constexpr size_t OFF_HS   = OFF_WOUT + (size_t)TT * 2 * HH * 2;     // bf16 [2][2][64,512]
static constexpr size_t OFF_EM   = OFF_HS   + (size_t)2 * 2 * BB * HH * 2; // f32 [M,32]
static constexpr size_t OFF_CTR  = OFF_EM   + (size_t)MM * TT * 4;         // u32

extern "C" void kernel_launch(void* const* d_in, const int* in_sizes, int n_in,
                              void* d_out, int out_size, void* d_ws, size_t ws_size,
                              hipStream_t stream) {
  const int*   sentence = (const int*)d_in[0];
  const int*   lengths  = (const int*)d_in[1];
  const int*   labels   = (const int*)d_in[2];
  // d_in[3] = mask (recomputed from lengths on device)
  const float* emb      = (const float*)d_in[4];
  const float* w_ih_f   = (const float*)d_in[5];
  const float* w_hh_f   = (const float*)d_in[6];
  const float* b_f      = (const float*)d_in[7];
  const float* w_ih_b   = (const float*)d_in[8];
  const float* w_hh_b   = (const float*)d_in[9];
  const float* b_b      = (const float*)d_in[10];
  const float* w_out    = (const float*)d_in[11];
  const float* b_out    = (const float*)d_in[12];
  const float* start_t  = (const float*)d_in[13];
  const float* end_t    = (const float*)d_in[14];
  const float* trans    = (const float*)d_in[15];

  char* ws = (char*)d_ws;
  float*        xw     = (float*)       (ws + OFF_XW);
  __bf16*       xb     = (__bf16*)      (ws + OFF_X);
  __bf16*       h_all  = (__bf16*)      (ws + OFF_HA);
  __bf16*       Wih    = (__bf16*)      (ws + OFF_WIH);
  __bf16*       Whh    = (__bf16*)      (ws + OFF_WHH);
  __bf16*       Wout   = (__bf16*)      (ws + OFF_WOUT);
  __bf16*       hstate = (__bf16*)      (ws + OFF_HS);
  float*        emis   = (float*)       (ws + OFF_EM);
  unsigned int* ctr    = (unsigned int*)(ws + OFF_CTR);
  float*        out    = (float*)d_out;

  // 0) init (zero output, barrier counter, h-state buffers)
  {
    const int nh = 2 * 2 * BB * HH;
    init_kernel<<<(nh + 255) / 256, 256, 0, stream>>>(out, ctr, hstate, nh);
  }
  // 1) weight converts to bf16 (layouts are already [N,K] row-major)
  {
    int n;
    n = G4 * DD; cvt_kernel<<<(n + 255) / 256, 256, 0, stream>>>(w_ih_f, Wih,            n);
    n = G4 * DD; cvt_kernel<<<(n + 255) / 256, 256, 0, stream>>>(w_ih_b, Wih + G4 * DD,  n);
    n = G4 * HH; cvt_kernel<<<(n + 255) / 256, 256, 0, stream>>>(w_hh_f, Whh,            n);
    n = G4 * HH; cvt_kernel<<<(n + 255) / 256, 256, 0, stream>>>(w_hh_b, Whh + G4 * HH,  n);
    n = TT * 2 * HH; cvt_kernel<<<(n + 255) / 256, 256, 0, stream>>>(w_out, Wout,        n);
  }
  // 2) embedding gather + convert
  {
    const int nthr = MM * (DD / 4);
    embed_kernel<<<(nthr + 255) / 256, 256, 0, stream>>>(sentence, emb, xb);
  }
  // 3) input projection for both directions (WMMA GEMM, bias fused)
  {
    const int waves  = (MM / 16) * 64;   // 131072 waves, 4 N-tiles each
    const int blocks = waves / 8;        // 256-thread blocks (8 waves)
    gemm_xw_kernel<<<blocks, 256, 0, stream>>>(xb, Wih, b_f, b_b, xw);
  }
  // 4) persistent bidirectional LSTM recurrence (Whh staged in 260 KB LDS)
  lstm_kernel<<<64, 128, LDS_BYTES, stream>>>(xw, Whh, lengths, hstate, h_all, ctr);
  // 5) emissions GEMM
  {
    const int waves  = MM / 16;          // 2048
    const int blocks = waves / 4;        // 128-thread blocks
    emis_kernel<<<blocks, 128, 0, stream>>>(h_all, Wout, b_out, emis);
  }
  // 6) CRF NLL, summed into out[0]
  crf_kernel<<<BB, 32, 0, stream>>>(emis, labels, lengths, start_t, end_t, trans, out);
}
